// Transformer_88751204204644
// MI455X (gfx1250) — matrix operand order repaired and checked by reading, not compiled, not run on hardware
//
#include <hip/hip_runtime.h>
#include <hip/hip_bf16.h>

// ---------------------------------------------------------------------------
// Types for CDNA5 WMMA (gfx1250, wave32)
// ---------------------------------------------------------------------------
typedef _Float16 half_t;
typedef __attribute__((ext_vector_type(16))) _Float16 v16h;
typedef __attribute__((ext_vector_type(8)))  _Float16 v8h;
typedef __attribute__((ext_vector_type(8)))  float    v8f;

union FragAB { v16h v; v8h h[2]; };
union FragC  { v8f  v; float f[8]; };

#define DEPTH 4
#define DIM   512
#define HEADS 8
#define DHEAD 64
#define MLP   2048
#define NTOK  1024           // tokens per path
#define BATCH 4
#define MROWS (BATCH * NTOK) // 4096 rows for all path GEMMs
#define SCALE 0.125f         // 64^-0.5
#define EPS_LN 1e-5f

enum { MODE_QKV = 1, MODE_GELU = 2, MODE_RESID = 3 };

// ---------------------------------------------------------------------------
// Fragment loaders
// A: row-major [M,K] fp16.  16x32 A-fragment at (row0, k0).
//   lane layout: l=lane&15 -> M=row0+l ; hi=lane>>4 selects K sub-blocks:
//   halfs 0..7  -> K = k0 + 8*hi + 0..7
//   halfs 8..15 -> K = k0 + 16 + 8*hi + 0..7
// BT: row-major [N,K] fp16 (i.e. logical B[K,N] column-major). 32x16 B-frag:
//   lane col N = n0 + l ; halfs 0..7 -> K = k0 + 8*hi + 0..7, halfs 8..15 -> K = k0 + 16 + 8*hi + 0..7
// ---------------------------------------------------------------------------
__device__ __forceinline__ FragAB load_a(const half_t* __restrict__ A, int row0,
                                         int k0, int lda, int l, int hi) {
  FragAB f;
  const half_t* p = A + (size_t)(row0 + l) * lda + k0 + 8 * hi;
  f.h[0] = *(const v8h*)p;
  f.h[1] = *(const v8h*)(p + 16);
  return f;
}

__device__ __forceinline__ FragAB load_bt(const half_t* __restrict__ BT, int k0,
                                          int n0, int ldbt, int l, int hi) {
  FragAB f;
  const half_t* p = BT + (size_t)(n0 + l) * ldbt + k0 + 8 * hi;
  f.h[0] = *(const v8h*)p;
  f.h[1] = *(const v8h*)(p + 16);
  return f;
}

__device__ __forceinline__ v8f wmma_f16(const FragAB& a, const FragAB& b, v8f c) {
  return __builtin_amdgcn_wmma_f32_16x16x32_f16(false, a.v, false, b.v,
                                                (short)0, c, false, false);
}

__device__ __forceinline__ float gelu_exact(float v) {
  return 0.5f * v * (1.0f + erff(v * 0.7071067811865476f));
}

// ---------------------------------------------------------------------------
// Weight convert + transpose: fp32 [K,N] -> fp16 [N,K], 16 tensors (layer*slot)
// ---------------------------------------------------------------------------
__global__ __launch_bounds__(256) void k_convT(const float* __restrict__ W,
                                               half_t* __restrict__ WT,
                                               int K, int N) {
  size_t tensor = blockIdx.y;
  size_t per = (size_t)K * N;
  size_t idx = (size_t)blockIdx.x * 256 + threadIdx.x;
  if (idx >= per) return;
  int k = (int)(idx / N);
  int n = (int)(idx % N);
  WT[tensor * per + (size_t)n * K + k] = (half_t)W[tensor * per + idx];
}

// ---------------------------------------------------------------------------
// LayerNorm over 512-wide rows of xbuf (fp32) -> fp16 [4096, 512]
// row m: b = m>>10, t = m&1023; xbuf row = b*2048 + path*1024 + t
// ---------------------------------------------------------------------------
__global__ __launch_bounds__(256) void k_ln(const float* __restrict__ x,
                                            const float* __restrict__ g,
                                            const float* __restrict__ b,
                                            half_t* __restrict__ out, int path) {
  __shared__ float sbuf[256];
  int row = blockIdx.x;
  int tid = threadIdx.x;
  int bb = row >> 10, t = row & 1023;
  const float* xr = x + ((size_t)(bb * 2048 + path * NTOK + t)) * DIM;
  float x0 = xr[tid], x1 = xr[tid + 256];

  sbuf[tid] = x0 + x1;
  __syncthreads();
  for (int st = 128; st > 0; st >>= 1) {
    if (tid < st) sbuf[tid] += sbuf[tid + st];
    __syncthreads();
  }
  float mean = sbuf[0] * (1.0f / DIM);
  __syncthreads();

  sbuf[tid] = x0 * x0 + x1 * x1;
  __syncthreads();
  for (int st = 128; st > 0; st >>= 1) {
    if (tid < st) sbuf[tid] += sbuf[tid + st];
    __syncthreads();
  }
  float var = sbuf[0] * (1.0f / DIM) - mean * mean;
  float rstd = rsqrtf(var + EPS_LN);

  half_t* orow = out + (size_t)row * DIM;
  orow[tid]       = (half_t)((x0 - mean) * rstd * g[tid] + b[tid]);
  orow[tid + 256] = (half_t)((x1 - mean) * rstd * g[tid + 256] + b[tid + 256]);
}

// ---------------------------------------------------------------------------
// WMMA GEMM: C[M,N] = A[M,K] (fp16 row-major) @ BT[N,K]^T (fp16)
// block = 256 threads = 8 waves; wave tile = 32x64 (2x4 fragments);
// block tile = 256x64.  grid = (M/256, N/64).
// Epilogues: QKV scatter, GELU->fp16, bias+residual->fp32 xbuf.
// ---------------------------------------------------------------------------
__global__ __launch_bounds__(256) void k_gemm(const half_t* __restrict__ A,
                                              const half_t* __restrict__ BT,
                                              const float* __restrict__ bias,
                                              int M, int N, int K, int mode,
                                              half_t* __restrict__ out16,
                                              float* __restrict__ resid,
                                              half_t* __restrict__ qb,
                                              half_t* __restrict__ kb,
                                              half_t* __restrict__ vtb) {
  int tid = threadIdx.x;
  int wave = tid >> 5;
  int lane = tid & 31;
  int l = lane & 15, hi = lane >> 4;
  int m0 = blockIdx.x * 256 + wave * 32;
  int n0 = blockIdx.y * 64;

  FragC acc[2][4];
#pragma unroll
  for (int mi = 0; mi < 2; ++mi)
#pragma unroll
    for (int nt = 0; nt < 4; ++nt)
#pragma unroll
      for (int i = 0; i < 8; ++i) acc[mi][nt].f[i] = 0.0f;

  for (int kk = 0; kk < K; kk += 32) {
    // prefetch next K tile of A into cache (global_prefetch_b8)
    if (kk + 32 < K) {
      __builtin_prefetch(A + (size_t)(m0 + l) * K + kk + 32, 0, 1);
      __builtin_prefetch(BT + (size_t)(n0 + l) * K + kk + 32, 0, 1);
    }
    FragAB a0 = load_a(A, m0, kk, K, l, hi);
    FragAB a1 = load_a(A, m0 + 16, kk, K, l, hi);
#pragma unroll
    for (int nt = 0; nt < 4; ++nt) {
      FragAB bt = load_bt(BT, kk, n0 + 16 * nt, K, l, hi);
      acc[0][nt].v = wmma_f16(a0, bt, acc[0][nt].v);
      acc[1][nt].v = wmma_f16(a1, bt, acc[1][nt].v);
    }
  }

  // epilogue: C-frag element (r) -> row m0+16*mi+8*hi+r, col n0+16*nt+l
#pragma unroll
  for (int mi = 0; mi < 2; ++mi) {
#pragma unroll
    for (int nt = 0; nt < 4; ++nt) {
      int c = n0 + 16 * nt + l;
#pragma unroll
      for (int r = 0; r < 8; ++r) {
        int m = m0 + 16 * mi + 8 * hi + r;
        float val = acc[mi][nt].f[r];
        if (mode == MODE_QKV) {
          int bb = m >> 10, t = m & 1023;
          int part = c >> 9;          // 0:q 1:k 2:v
          int cc = c & 511;
          int h = cc >> 6, d = cc & 63;
          if (part == 0) {
            qb[((size_t)(bb * HEADS + h) * NTOK + t) * DHEAD + d] =
                (half_t)(val * SCALE);
          } else if (part == 1) {
            kb[((size_t)(bb * HEADS + h) * NTOK + t) * DHEAD + d] = (half_t)val;
          } else {
            vtb[((size_t)(bb * HEADS + h) * DHEAD + d) * NTOK + t] = (half_t)val;
          }
        } else if (mode == MODE_GELU) {
          out16[(size_t)m * N + c] = (half_t)gelu_exact(val + bias[c]);
        } else { // MODE_RESID: bias + residual add into fp32 xbuf (path base)
          int bb = m >> 10, t = m & 1023;
          float* rp = resid + ((size_t)(bb * 2048 + t)) * DIM + c;
          *rp = *rp + val + bias[c];
        }
      }
    }
  }
}

// ---------------------------------------------------------------------------
// Fused flash-attention per (b,h): Q[n,64], K[n,64], VT[64,n] -> AO[b,n,512]
// 4 waves/block; each wave owns 16 query rows; online softmax over j-tiles
// of 32; P bounced through LDS to convert C-layout -> A-layout.
// ---------------------------------------------------------------------------
__global__ __launch_bounds__(128) void k_attn(const half_t* __restrict__ Q,
                                              const half_t* __restrict__ Km,
                                              const half_t* __restrict__ VT,
                                              half_t* __restrict__ AO, int n) {
  __shared__ alignas(16) half_t plds[4][16][32];

  int bh = blockIdx.x;
  int b = bh >> 3, h = bh & 7;
  const half_t* Qp = Q + (size_t)bh * n * DHEAD;
  const half_t* Kp = Km + (size_t)bh * n * DHEAD;
  const half_t* Vp = VT + (size_t)bh * DHEAD * n;
  half_t* AOp = AO + (size_t)b * n * (HEADS * DHEAD) + h * DHEAD;

  int tid = threadIdx.x;
  int wave = tid >> 5;
  int lane = tid & 31;
  int l = lane & 15, hi = lane >> 4;
  int i0 = (blockIdx.y * 4 + wave) * 16;

  // Q fragments: [16 x 64] = 2 A-fragments (K = 0..31, 32..63), pre-scaled
  FragAB qf[2];
#pragma unroll
  for (int fi = 0; fi < 2; ++fi) qf[fi] = load_a(Qp, i0, 32 * fi, DHEAD, l, hi);

  float mrow[8], lrow[8];
  FragC o[4];
#pragma unroll
  for (int r = 0; r < 8; ++r) { mrow[r] = -1e30f; lrow[r] = 0.0f; }
#pragma unroll
  for (int nt = 0; nt < 4; ++nt)
#pragma unroll
    for (int i = 0; i < 8; ++i) o[nt].f[i] = 0.0f;

  for (int j0 = 0; j0 < n; j0 += 32) {
    // scores for 32 key columns: S[i, j0..j0+31], full 64-d dot
    FragC s0, s1;
#pragma unroll
    for (int i = 0; i < 8; ++i) { s0.f[i] = 0.0f; s1.f[i] = 0.0f; }
#pragma unroll
    for (int fi = 0; fi < 2; ++fi) {
      FragAB bk0 = load_bt(Kp, 32 * fi, j0, DHEAD, l, hi);
      FragAB bk1 = load_bt(Kp, 32 * fi, j0 + 16, DHEAD, l, hi);
      s0.v = wmma_f16(qf[fi], bk0, s0.v);
      s1.v = wmma_f16(qf[fi], bk1, s1.v);
    }

    // online softmax; C-frag row m = 8*hi + r lives across the 16-lane half
#pragma unroll
    for (int r = 0; r < 8; ++r) {
      float v0 = s0.f[r], v1 = s1.f[r];
      float mx = fmaxf(v0, v1);
#pragma unroll
      for (int off = 8; off > 0; off >>= 1) mx = fmaxf(mx, __shfl_xor(mx, off, 16));
      float mn = fmaxf(mrow[r], mx);
      float sc = __expf(mrow[r] - mn);
      mrow[r] = mn;
      float p0 = __expf(v0 - mn);
      float p1 = __expf(v1 - mn);
      float ps = p0 + p1;
#pragma unroll
      for (int off = 8; off > 0; off >>= 1) ps += __shfl_xor(ps, off, 16);
      lrow[r] = lrow[r] * sc + ps;
#pragma unroll
      for (int nt = 0; nt < 4; ++nt) o[nt].f[r] *= sc;
      plds[wave][8 * hi + r][l] = (half_t)p0;
      plds[wave][8 * hi + r][16 + l] = (half_t)p1;
    }
    asm volatile("s_wait_dscnt 0x0" ::: "memory");

    // P A-fragment from LDS (row l, halfs k = 8*hi.., 16+8*hi..)
    FragAB pa;
    pa.h[0] = *(const v8h*)&plds[wave][l][8 * hi];
    pa.h[1] = *(const v8h*)&plds[wave][l][16 + 8 * hi];

    // O[i, 0..63] += P[i, j0..j0+31] @ V[j0..j0+31, 0..63] (V via VT rows)
#pragma unroll
    for (int nt = 0; nt < 4; ++nt) {
      FragAB bv = load_bt(Vp, j0, 16 * nt, n, l, hi);
      o[nt].v = wmma_f16(pa, bv, o[nt].v);
    }
  }

  // normalize and store fp16 (concat heads layout [n, 512])
#pragma unroll
  for (int nt = 0; nt < 4; ++nt) {
#pragma unroll
    for (int r = 0; r < 8; ++r) {
      float val = o[nt].f[r] / lrow[r];
      AOp[(size_t)(i0 + 8 * hi + r) * (HEADS * DHEAD) + 16 * nt + l] = (half_t)val;
    }
  }
}

// ---------------------------------------------------------------------------
// Final gather: d_out = concat(x[:, :1024, :], x[:, 1024:, :]) flat, fp32
// ---------------------------------------------------------------------------
__global__ __launch_bounds__(256) void k_final(const float* __restrict__ x,
                                               float* __restrict__ out) {
  size_t idx = (size_t)blockIdx.x * 256 + threadIdx.x;
  // 2 * 4 * 1024 * 512 = 4194304 elements total
  int path = (int)(idx >> 21);
  size_t rem = idx & ((1u << 21) - 1);
  int b = (int)(rem >> 19);
  size_t r2 = rem & ((1u << 19) - 1);
  int t = (int)(r2 >> 9);
  int c = (int)(r2 & 511);
  out[idx] = x[((size_t)(b * 2048 + path * NTOK + t)) * DIM + c];
}

// ---------------------------------------------------------------------------
// Host orchestration
// ---------------------------------------------------------------------------
extern "C" void kernel_launch(void* const* d_in, const int* in_sizes, int n_in,
                              void* d_out, int out_size, void* d_ws, size_t ws_size,
                              hipStream_t stream) {
  (void)in_sizes; (void)n_in; (void)out_size; (void)ws_size;

  const float* x_in  = (const float*)d_in[0];
  const float* ln_g  = (const float*)d_in[1];
  const float* ln_b  = (const float*)d_in[2];
  const float* qkv_w = (const float*)d_in[3];
  const float* out_w = (const float*)d_in[4];
  const float* out_b = (const float*)d_in[5];
  const float* ff_w1 = (const float*)d_in[6];
  const float* ff_b1 = (const float*)d_in[7];
  const float* ff_w2 = (const float*)d_in[8];
  const float* ff_b2 = (const float*)d_in[9];

  char* base = (char*)d_ws;
  size_t off = 0;
  auto alloc = [&](size_t bytes) -> void* {
    void* p = base + off;
    off += (bytes + 255) & ~(size_t)255;
    return p;
  };

  float*  xbuf  = (float*)alloc((size_t)BATCH * 2048 * DIM * sizeof(float));
  half_t* qkvT  = (half_t*)alloc((size_t)16 * DIM * (3 * DIM) * sizeof(half_t));
  half_t* outT  = (half_t*)alloc((size_t)16 * DIM * DIM * sizeof(half_t));
  half_t* ff1T  = (half_t*)alloc((size_t)16 * DIM * MLP * sizeof(half_t));
  half_t* ff2T  = (half_t*)alloc((size_t)16 * MLP * DIM * sizeof(half_t));
  half_t* h16[2], *Qb[2], *Kb[2], *VTb[2], *AO[2], *mid[2];
  for (int p = 0; p < 2; ++p) {
    h16[p] = (half_t*)alloc((size_t)MROWS * DIM * sizeof(half_t));
    Qb[p]  = (half_t*)alloc((size_t)BATCH * HEADS * NTOK * DHEAD * sizeof(half_t));
    Kb[p]  = (half_t*)alloc((size_t)BATCH * HEADS * NTOK * DHEAD * sizeof(half_t));
    VTb[p] = (half_t*)alloc((size_t)BATCH * HEADS * DHEAD * NTOK * sizeof(half_t));
    AO[p]  = (half_t*)alloc((size_t)MROWS * DIM * sizeof(half_t));
    mid[p] = (half_t*)alloc((size_t)MROWS * MLP * sizeof(half_t));
  }

  hipMemcpyAsync(xbuf, x_in, (size_t)BATCH * 2048 * DIM * sizeof(float),
                 hipMemcpyDeviceToDevice, stream);

  // weight conversion (fp32 [K,N] -> fp16 [N,K]) for all 16 (layer,slot)
  k_convT<<<dim3((DIM * 3 * DIM + 255) / 256, 16), 256, 0, stream>>>(qkv_w, qkvT, DIM, 3 * DIM);
  k_convT<<<dim3((DIM * DIM + 255) / 256, 16), 256, 0, stream>>>(out_w, outT, DIM, DIM);
  k_convT<<<dim3((DIM * MLP + 255) / 256, 16), 256, 0, stream>>>(ff_w1, ff1T, DIM, MLP);
  k_convT<<<dim3((MLP * DIM + 255) / 256, 16), 256, 0, stream>>>(ff_w2, ff2T, MLP, DIM);

  const dim3 gQKV(MROWS / 256, (3 * DIM) / 64);
  const dim3 gOUT(MROWS / 256, DIM / 64);
  const dim3 gFF1(MROWS / 256, MLP / 64);
  const dim3 gATT(BATCH * HEADS, NTOK / 64);

  auto ln_launch = [&](int l, int slot, int p) {
    k_ln<<<MROWS, 256, 0, stream>>>(xbuf, ln_g + (size_t)(l * 8 + slot) * DIM,
                                    ln_b + (size_t)(l * 8 + slot) * DIM, h16[p], p);
  };
  auto qkv_launch = [&](int l, int s, int p) {
    k_gemm<<<gQKV, 256, 0, stream>>>(h16[p], qkvT + (size_t)(l * 4 + s) * (3 * DIM) * DIM,
                                     nullptr, MROWS, 3 * DIM, DIM, MODE_QKV,
                                     nullptr, nullptr, Qb[p], Kb[p], VTb[p]);
  };
  auto outproj_launch = [&](int l, int s, int p) {
    k_gemm<<<gOUT, 256, 0, stream>>>(AO[p], outT + (size_t)(l * 4 + s) * DIM * DIM,
                                     out_b + (size_t)(l * 4 + s) * DIM,
                                     MROWS, DIM, DIM, MODE_RESID, nullptr,
                                     xbuf + (size_t)p * NTOK * DIM,
                                     nullptr, nullptr, nullptr);
  };
  auto ffn_launch = [&](int l, int s, int p) {
    k_gemm<<<gFF1, 256, 0, stream>>>(h16[p], ff1T + (size_t)(l * 4 + s) * MLP * DIM,
                                     ff_b1 + (size_t)(l * 4 + s) * MLP,
                                     MROWS, MLP, DIM, MODE_GELU, mid[p],
                                     nullptr, nullptr, nullptr, nullptr);
    k_gemm<<<gOUT, 256, 0, stream>>>(mid[p], ff2T + (size_t)(l * 4 + s) * DIM * MLP,
                                     ff_b2 + (size_t)(l * 4 + s) * DIM,
                                     MROWS, DIM, MLP, MODE_RESID, nullptr,
                                     xbuf + (size_t)p * NTOK * DIM,
                                     nullptr, nullptr, nullptr);
  };

  for (int l = 0; l < DEPTH; ++l) {
    // ---- dual self-attention (LN slots 0/1, weight slots 0/1)
    for (int p = 0; p < 2; ++p) { ln_launch(l, p, p); qkv_launch(l, p, p); }
    k_attn<<<gATT, 128, 0, stream>>>(Qb[0], Kb[0], VTb[0], AO[0], NTOK);
    k_attn<<<gATT, 128, 0, stream>>>(Qb[1], Kb[1], VTb[1], AO[1], NTOK);
    for (int p = 0; p < 2; ++p) outproj_launch(l, p, p);

    // ---- dual FFN 1 (LN slots 2/3, weight slots 0/1)
    for (int p = 0; p < 2; ++p) { ln_launch(l, 2 + p, p); ffn_launch(l, p, p); }

    // ---- cross-attention (LN slots 4/5, qkv/out weight slots 2/3)
    for (int p = 0; p < 2; ++p) { ln_launch(l, 4 + p, p); qkv_launch(l, 2 + p, p); }
    k_attn<<<gATT, 128, 0, stream>>>(Qb[0], Kb[1], VTb[1], AO[0], NTOK); // o1 = sdpa(q1,k2,v2)
    k_attn<<<gATT, 128, 0, stream>>>(Qb[1], Kb[0], VTb[0], AO[1], NTOK); // o2 = sdpa(q2,k1,v1)
    for (int p = 0; p < 2; ++p) outproj_launch(l, 2 + p, p);

    // ---- dual FFN 2 (LN slots 6/7, weight slots 2/3)
    for (int p = 0; p < 2; ++p) { ln_launch(l, 6 + p, p); ffn_launch(l, 2 + p, p); }
  }

  // gather the two path halves into d_out (fp32)
  k_final<<<(2u * BATCH * NTOK * DIM) / 256, 256, 0, stream>>>(xbuf, (float*)d_out);
}
